// JEM_84542136254639
// MI455X (gfx1250) — matrix operand order repaired and checked by reading, not compiled, not run on hardware
//
#include <hip/hip_runtime.h>

// ---------------------------------------------------------------------------
// JEM / SGLD sampler for MI455X (gfx1250), bf16 WMMA core, f32 accumulate,
// async global->LDS double-buffered staging.
// ---------------------------------------------------------------------------

typedef __bf16 bf16;
typedef bf16  v16bf __attribute__((ext_vector_type(16)));
typedef bf16  v8bf  __attribute__((ext_vector_type(8)));
typedef float v8f   __attribute__((ext_vector_type(8)));
typedef float v4f   __attribute__((ext_vector_type(4)));
typedef int   v4i   __attribute__((ext_vector_type(4)));

#define BUFSZ   10000
#define FEAT    256
#define HID     512
#define NCLS    10
#define BSZ     8192
#define LDN     40
#define NPAD    64      // logits padded to 64 cols (tile width)
#define KPAD    32      // dlogits padded to 32 K   (WMMA K)
#define REINIT  0.05f
#define LR      1.0f
#define SGSTD   0.01f

// GEMM tiling
#define TM 128
#define TN 64
#define TK 32
#define AS_STRIDE 36    // floats per LDS row of A tile (pad -> conflict-free)
#define BS_STRIDE 40    // bf16 per LDS row of Bt tile (pad -> conflict-free)
#define CHUNK_OPS 5     // async b128 issues per thread per K-chunk (4 A + 1 B)

// Epilogue modes
#define EP_BIAS      0   // C = acc + bias
#define EP_BIAS_RELU 1   // C = relu(acc + bias)
#define EP_MASK      2   // C = (mask>0) ? acc : 0
#define EP_SGLD      3   // C = xin - LR*acc + SGSTD*noise

// ---- gfx1250 async global->LDS support detection --------------------------
#if defined(__has_builtin)
#  if __has_builtin(__builtin_amdgcn_global_load_async_to_lds_b128)
#    define USE_ASYNC_LDS 1
#  endif
#endif
#ifndef USE_ASYNC_LDS
#  define USE_ASYNC_LDS 0
#endif

#if defined(__has_builtin)
#  if __has_builtin(__builtin_amdgcn_s_wait_asynccnt)
#    define HAVE_WAIT_ASYNC_BUILTIN 1
#  endif
#endif
#ifndef HAVE_WAIT_ASYNC_BUILTIN
#  define HAVE_WAIT_ASYNC_BUILTIN 0
#endif

__device__ __forceinline__ void async_copy_b128(const void* g, void* l) {
#if USE_ASYNC_LDS
    __builtin_amdgcn_global_load_async_to_lds_b128(
        (__attribute__((address_space(1))) v4i*)const_cast<void*>(g),
        (__attribute__((address_space(3))) v4i*)l, 0, 0);
#else
    (void)g; (void)l;
#endif
}

template <int N>
__device__ __forceinline__ void wait_async() {
#if USE_ASYNC_LDS
#  if HAVE_WAIT_ASYNC_BUILTIN
    __builtin_amdgcn_s_wait_asynccnt(N);
#  else
    asm volatile("s_wait_asynccnt %0" :: "n"(N) : "memory");
#  endif
#endif
}

// ---------------------------------------------------------------------------
// Tiled WMMA GEMM: C[M,N] = A[M,K](f32) * B[K,N](bf16) (+ epilogue)
// B is passed TRANSPOSED (N-major): BT[n*K + k] == B[k][n].
// M % 128 == 0, N % 64 == 0, K % 32 == 0 (guaranteed by padding).
// block = 256 threads = 8 waves; wave w computes rows 16w..16w+15, cols 0..63.
// K-chunks are double-buffered in LDS via GLOBAL_LOAD_ASYNC_TO_LDS_B128.
// ---------------------------------------------------------------------------
template <int MODE>
__global__ __launch_bounds__(256) void jem_gemm_wmma(
    const float* __restrict__ A, const bf16* __restrict__ BT,
    float* __restrict__ C,
    const float* __restrict__ bias,   // EP_BIAS / EP_BIAS_RELU
    const float* __restrict__ mask,   // EP_MASK
    const float* __restrict__ xin,    // EP_SGLD
    const float* __restrict__ noise,  // EP_SGLD
    int M, int N, int K)
{
    __shared__ float As[2][TM * AS_STRIDE];     // 2 x 18.0 KB
    __shared__ bf16  Bt[2][TN * BS_STRIDE];     // 2 x  5.0 KB  (Bt[n][k])

    const int t    = threadIdx.x;
    const int wave = t >> 5;
    const int lane = t & 31;
    const int half = lane >> 4;     // K-half selector for A/B fragments
    const int l16  = lane & 15;

    const int blockRow = blockIdx.x * TM;
    const int blockCol = blockIdx.y * TN;

    // staging index precompute
    const int a_col4 = (t & 7) * 4;   // float4 column within K-chunk
    const int a_row0 = t >> 3;        // 0..31, +32 per pass
    const int b_n    = t >> 2;        // 0..63 : output column within tile
    const int b_ks   = (t & 3) * 8;   // 8 consecutive k per thread

    v8f acc[4];
#pragma unroll
    for (int c = 0; c < 4; ++c)
#pragma unroll
        for (int r = 0; r < 8; ++r) acc[c][r] = 0.0f;

    const int nChunks = K / TK;

    // ---- stage one K-chunk into LDS buffer `buf` ----
    auto stage = [&](int chunk, int buf) {
        const int k0 = chunk * TK;
        float* AsB = As[buf];
        bf16*  BtB = Bt[buf];
#if USE_ASYNC_LDS
#pragma unroll
        for (int p = 0; p < 4; ++p) {
            const int row = a_row0 + 32 * p;
            async_copy_b128(A + (size_t)(blockRow + row) * K + k0 + a_col4,
                            AsB + row * AS_STRIDE + a_col4);
        }
        async_copy_b128(BT + (size_t)(blockCol + b_n) * K + k0 + b_ks,
                        BtB + b_n * BS_STRIDE + b_ks);
#else
        v4f at[4];
#pragma unroll
        for (int p = 0; p < 4; ++p)
            at[p] = *(const v4f*)(A + (size_t)(blockRow + a_row0 + 32 * p) * K + k0 + a_col4);
        v8bf bt = *(const v8bf*)(BT + (size_t)(blockCol + b_n) * K + k0 + b_ks);
#pragma unroll
        for (int p = 0; p < 4; ++p)
            *(v4f*)(AsB + (a_row0 + 32 * p) * AS_STRIDE + a_col4) = at[p];
        *(v8bf*)(BtB + b_n * BS_STRIDE + b_ks) = bt;
#endif
    };

    stage(0, 0);                       // prologue: chunk 0 -> buffer 0

    for (int c = 0; c < nChunks; ++c) {
        const int cur = c & 1;
        if (c + 1 < nChunks) {
            stage(c + 1, cur ^ 1);     // overlap next chunk's DMA with compute
            wait_async<CHUNK_OPS>();   // chunk c's copies retired (in-order)
        } else {
            wait_async<0>();
        }
        __syncthreads();               // all waves' chunk-c data visible

        // ---- A fragment: ISA 16-bit A layout.
        // lanes 0-15 (half=0): K = {0..7, 16..23}; lanes 16-31: K = {8..15, 24..31}
        v16bf afrag;
        {
            const float* ap = As[cur] + (wave * 16 + l16) * AS_STRIDE;
#pragma unroll
            for (int j = 0; j < 8; ++j) afrag[j]     = (bf16)ap[half * 8 + j];
#pragma unroll
            for (int j = 0; j < 8; ++j) afrag[8 + j] = (bf16)ap[16 + half * 8 + j];
        }
        // ---- B fragments + WMMA: lane holds column n=l16, K = {half*8+0..7, 16+half*8+0..7} ----
#pragma unroll
        for (int cc = 0; cc < 4; ++cc) {
            const bf16* bp = Bt[cur] + (cc * 16 + l16) * BS_STRIDE;
            v16bf bfrag;
#pragma unroll
            for (int j = 0; j < 16; ++j) bfrag[j] = bp[(j & 8) * 2 + half * 8 + (j & 7)];
            acc[cc] = __builtin_amdgcn_wmma_f32_16x16x32_bf16(
                false, afrag, false, bfrag, (short)0, acc[cc], false, false);
        }
        __syncthreads();               // done reading buffer `cur`
    }

    // ---- epilogue: D layout, VGPR r -> row (half*8 + r), lane l16 -> col ----
#pragma unroll
    for (int c = 0; c < 4; ++c) {
#pragma unroll
        for (int r = 0; r < 8; ++r) {
            const int row = blockRow + wave * 16 + half * 8 + r;
            const int col = blockCol + c * 16 + l16;
            const size_t idx = (size_t)row * N + col;
            float v = acc[c][r];
            if (MODE == EP_BIAS)      { v += bias[col]; }
            if (MODE == EP_BIAS_RELU) { v += bias[col]; v = v > 0.0f ? v : 0.0f; }
            if (MODE == EP_MASK)      { v = (mask[idx] > 0.0f) ? v : 0.0f; }
            if (MODE == EP_SGLD)      { v = xin[idx] - LR * v + SGSTD * noise[idx]; }
            C[idx] = v;
        }
    }
}

// ---------------------------------------------------------------------------
// Small one-time prep / elementwise kernels
// ---------------------------------------------------------------------------
__global__ void k_cvt_bf16(const float* __restrict__ s, bf16* __restrict__ d, int n) {
    int i = blockIdx.x * blockDim.x + threadIdx.x;
    if (i < n) d[i] = (bf16)s[i];
}

// dst[N,K] (bf16) = transpose of src[K,N] (f32)
__global__ void k_transpose_bf16(const float* __restrict__ s, bf16* __restrict__ d, int K, int N) {
    int i = blockIdx.x * blockDim.x + threadIdx.x;
    if (i < K * N) { int n = i / K, k = i % K; d[(size_t)n * K + k] = (bf16)s[(size_t)k * N + n]; }
}

// BT for logits GEMM: [NPAD,HID]; BT[n][k] = (n<NCLS) ? W3[k][n] : 0
__global__ void k_w3_padT(const float* __restrict__ s, bf16* __restrict__ d) {
    int i = blockIdx.x * blockDim.x + threadIdx.x;
    if (i < NPAD * HID) { int n = i / HID, k = i % HID;
        d[i] = (n < NCLS) ? (bf16)s[k * NCLS + n] : (bf16)0.0f; }
}

// BT for dh2 GEMM (B = W3^T [KPAD,HID]): [HID,KPAD]; BT[n][k] = (k<NCLS) ? W3[n][k] : 0
__global__ void k_w3tT(const float* __restrict__ s, bf16* __restrict__ d) {
    int i = blockIdx.x * blockDim.x + threadIdx.x;
    if (i < HID * KPAD) { int n = i / KPAD, k = i % KPAD;
        d[i] = (k < NCLS) ? (bf16)s[n * NCLS + k] : (bf16)0.0f; }
}

__global__ void k_b3_pad(const float* __restrict__ s, float* __restrict__ d) {
    int i = blockIdx.x * blockDim.x + threadIdx.x;
    if (i < NPAD) d[i] = (i < NCLS) ? s[i] : 0.0f;
}

// sample_p_0: mix buffer rows with fresh uniform(-1,1)
__global__ void k_init_x0(const float* __restrict__ buffer, const float* __restrict__ u_rand,
                          const float* __restrict__ u_choose, const int* __restrict__ inds,
                          float* __restrict__ x) {
    int i = blockIdx.x * blockDim.x + threadIdx.x;
    if (i < BSZ * FEAT) {
        int row = i >> 8, f = i & (FEAT - 1);
        float rs = 2.0f * u_rand[i] - 1.0f;
        float bs = buffer[(size_t)inds[row] * FEAT + f];
        x[i] = (u_choose[row] < REINIT) ? rs : bs;
    }
}

// dE/dlogits = -softmax(logits); writes padded [BSZ,KPAD] (zeros beyond NCLS)
__global__ void k_softmax_grad(const float* __restrict__ logits, float* __restrict__ dlog) {
    int row = blockIdx.x * blockDim.x + threadIdx.x;
    if (row >= BSZ) return;
    const float* l = logits + (size_t)row * NPAD;
    float m = l[0];
#pragma unroll
    for (int c = 1; c < NCLS; ++c) m = l[c] > m ? l[c] : m;
    float s = 0.0f;
    float e[NCLS];
#pragma unroll
    for (int c = 0; c < NCLS; ++c) { e[c] = __expf(l[c] - m); s += e[c]; }
    float inv = 1.0f / s;
    float* d = dlog + (size_t)row * KPAD;
#pragma unroll
    for (int c = 0; c < NCLS; ++c) d[c] = -e[c] * inv;
#pragma unroll
    for (int c = NCLS; c < KPAD; ++c) d[c] = 0.0f;
}

__global__ void k_copy_f32(const float* __restrict__ s, float* __restrict__ d, int n) {
    int i = blockIdx.x * blockDim.x + threadIdx.x;
    if (i < n) d[i] = s[i];
}

// new_buffer[inds[i]] = xk[i]
__global__ void k_scatter(const float* __restrict__ xk, const int* __restrict__ inds,
                          float* __restrict__ obuf) {
    int i = blockIdx.x * blockDim.x + threadIdx.x;
    if (i < BSZ * FEAT) {
        int row = i >> 8, f = i & (FEAT - 1);
        obuf[(size_t)inds[row] * FEAT + f] = xk[i];
    }
}

// ---------------------------------------------------------------------------
// Workspace layout (bytes, 256-aligned)
// ---------------------------------------------------------------------------
#define ALIGN256(x) (((x) + 255) & ~(size_t)255)
static const size_t OFF_XA    = 0;
static const size_t OFF_XB    = OFF_XA    + ALIGN256((size_t)BSZ * FEAT * 4);
static const size_t OFF_H1    = OFF_XB    + ALIGN256((size_t)BSZ * FEAT * 4);
static const size_t OFF_H2    = OFF_H1    + ALIGN256((size_t)BSZ * HID * 4);
static const size_t OFF_DH2   = OFF_H2    + ALIGN256((size_t)BSZ * HID * 4);
static const size_t OFF_DH1   = OFF_DH2   + ALIGN256((size_t)BSZ * HID * 4);
static const size_t OFF_LOG   = OFF_DH1   + ALIGN256((size_t)BSZ * HID * 4);
static const size_t OFF_DLOG  = OFF_LOG   + ALIGN256((size_t)BSZ * NPAD * 4);
static const size_t OFF_W1B   = OFF_DLOG  + ALIGN256((size_t)BSZ * KPAD * 4);
static const size_t OFF_W1T   = OFF_W1B   + ALIGN256((size_t)FEAT * HID * 2);
static const size_t OFF_W2B   = OFF_W1T   + ALIGN256((size_t)FEAT * HID * 2);
static const size_t OFF_W2T   = OFF_W2B   + ALIGN256((size_t)HID * HID * 2);
static const size_t OFF_W3PT  = OFF_W2T   + ALIGN256((size_t)HID * HID * 2);
static const size_t OFF_W3TT  = OFF_W3PT  + ALIGN256((size_t)NPAD * HID * 2);
static const size_t OFF_B3P   = OFF_W3TT  + ALIGN256((size_t)HID * KPAD * 2);

extern "C" void kernel_launch(void* const* d_in, const int* in_sizes, int n_in,
                              void* d_out, int out_size, void* d_ws, size_t ws_size,
                              hipStream_t stream) {
    const float* buffer   = (const float*)d_in[0];
    const float* W1       = (const float*)d_in[1];
    const float* b1       = (const float*)d_in[2];
    const float* W2       = (const float*)d_in[3];
    const float* b2       = (const float*)d_in[4];
    const float* W3       = (const float*)d_in[5];
    const float* b3       = (const float*)d_in[6];
    const float* noise    = (const float*)d_in[7];
    const float* u_rand   = (const float*)d_in[8];
    const float* u_choose = (const float*)d_in[9];
    const int*   inds     = (const int*)d_in[10];

    char* w = (char*)d_ws;
    float* xA   = (float*)(w + OFF_XA);
    float* xB   = (float*)(w + OFF_XB);
    float* h1   = (float*)(w + OFF_H1);
    float* h2   = (float*)(w + OFF_H2);
    float* dh2  = (float*)(w + OFF_DH2);
    float* dh1  = (float*)(w + OFF_DH1);
    float* lg   = (float*)(w + OFF_LOG);
    float* dlog = (float*)(w + OFF_DLOG);
    bf16*  W1b  = (bf16*)(w + OFF_W1B);   // [FEAT,HID]  = BT for dx GEMM
    bf16*  W1t  = (bf16*)(w + OFF_W1T);   // [HID,FEAT]  = BT for fwd1 GEMM
    bf16*  W2b  = (bf16*)(w + OFF_W2B);   // [HID,HID]   = BT for dh1 GEMM
    bf16*  W2t  = (bf16*)(w + OFF_W2T);   // [HID,HID]   = BT for fwd2 GEMM
    bf16*  W3pT = (bf16*)(w + OFF_W3PT);  // [NPAD,HID]  = BT for logits GEMM
    bf16*  W3tT = (bf16*)(w + OFF_W3TT);  // [HID,KPAD]  = BT for dh2 GEMM
    float* b3p  = (float*)(w + OFF_B3P);

    const int T = 256;
    // ---- one-time weight prep (all GEMM B operands stored N-major) ----
    k_cvt_bf16<<<(FEAT * HID + T - 1) / T, T, 0, stream>>>(W1, W1b, FEAT * HID);
    k_transpose_bf16<<<(FEAT * HID + T - 1) / T, T, 0, stream>>>(W1, W1t, FEAT, HID);
    k_cvt_bf16<<<(HID * HID + T - 1) / T, T, 0, stream>>>(W2, W2b, HID * HID);
    k_transpose_bf16<<<(HID * HID + T - 1) / T, T, 0, stream>>>(W2, W2t, HID, HID);
    k_w3_padT<<<(NPAD * HID + T - 1) / T, T, 0, stream>>>(W3, W3pT);
    k_w3tT<<<(HID * KPAD + T - 1) / T, T, 0, stream>>>(W3, W3tT);
    k_b3_pad<<<1, NPAD, 0, stream>>>(b3, b3p);

    // ---- x0 ----
    k_init_x0<<<(BSZ * FEAT + T - 1) / T, T, 0, stream>>>(buffer, u_rand, u_choose, inds, xA);

    // ---- 40 SGLD steps ----
    float* xp[2] = { xA, xB };
    for (int s = 0; s < LDN; ++s) {
        float* xi = xp[s & 1];
        float* xo = xp[(s + 1) & 1];
        const float* nz = noise + (size_t)s * BSZ * FEAT;

        // h1 = relu(x @ W1 + b1)
        jem_gemm_wmma<EP_BIAS_RELU><<<dim3(BSZ / TM, HID / TN), T, 0, stream>>>(
            xi, W1t, h1, b1, nullptr, nullptr, nullptr, BSZ, HID, FEAT);
        // h2 = relu(h1 @ W2 + b2)
        jem_gemm_wmma<EP_BIAS_RELU><<<dim3(BSZ / TM, HID / TN), T, 0, stream>>>(
            h1, W2t, h2, b2, nullptr, nullptr, nullptr, BSZ, HID, HID);
        // logits(padded) = h2 @ W3p + b3p
        jem_gemm_wmma<EP_BIAS><<<dim3(BSZ / TM, NPAD / TN), T, 0, stream>>>(
            h2, W3pT, lg, b3p, nullptr, nullptr, nullptr, BSZ, NPAD, HID);
        // dlogits = -softmax(logits)  (K-padded)
        k_softmax_grad<<<(BSZ + T - 1) / T, T, 0, stream>>>(lg, dlog);
        // dh2 = (dlogits @ W3^T) * (h2 > 0)
        jem_gemm_wmma<EP_MASK><<<dim3(BSZ / TM, HID / TN), T, 0, stream>>>(
            dlog, W3tT, dh2, nullptr, h2, nullptr, nullptr, BSZ, HID, KPAD);
        // dh1 = (dh2 @ W2^T) * (h1 > 0)
        jem_gemm_wmma<EP_MASK><<<dim3(BSZ / TM, HID / TN), T, 0, stream>>>(
            dh2, W2b, dh1, nullptr, h1, nullptr, nullptr, BSZ, HID, HID);
        // x' = x - LR * (dh1 @ W1^T) + SGSTD * noise[s]   (fused SGLD update)
        jem_gemm_wmma<EP_SGLD><<<dim3(BSZ / TM, FEAT / TN), T, 0, stream>>>(
            dh1, W1b, xo, nullptr, nullptr, xi, nz, BSZ, FEAT, HID);
    }
    float* xk = xp[LDN & 1];  // LDN even -> xA

    // ---- outputs: [xk | new_buffer] ----
    float* out_xk  = (float*)d_out;
    float* out_buf = (float*)d_out + (size_t)BSZ * FEAT;
    k_copy_f32<<<(BSZ * FEAT + T - 1) / T, T, 0, stream>>>(xk, out_xk, BSZ * FEAT);
    k_copy_f32<<<(BUFSZ * FEAT + T - 1) / T, T, 0, stream>>>(buffer, out_buf, BUFSZ * FEAT);
    k_scatter<<<(BSZ * FEAT + T - 1) / T, T, 0, stream>>>(xk, inds, out_buf);
}
